// SelfAttention_47459388621125
// MI455X (gfx1250) — compile-verified, matrix operand order checked by reading, not run on hardware
//
#include <hip/hip_runtime.h>
#include <hip/hip_bf16.h>

typedef __attribute__((ext_vector_type(16))) __bf16 v16bf;
typedef __attribute__((ext_vector_type(8)))  __bf16 v8bf;
typedef __attribute__((ext_vector_type(8)))  float  v8f;

#define N_  4
#define L_  2048
#define E_  512
#define H_  8
#define D_  64
#define NL_ (N_ * L_)
#define NLE_ ((size_t)NL_ * E_)
#define LDP 72  // padded LDS row stride in bf16 elements (144B: 16B-aligned, bank-conflict-free)

// LDS byte-offset of a generic pointer known to point into LDS (addrspacecast -> AS3).
__device__ __forceinline__ uint32_t lds_offset(const void* p) {
  return (uint32_t)(size_t)(__attribute__((address_space(3))) const void*)p;
}

// CDNA5 async copy: 16B per lane, global -> LDS, tracked by ASYNCcnt.
__device__ __forceinline__ void async_load_b128(uint32_t lds_addr, const void* gptr) {
  asm volatile("global_load_async_to_lds_b128 %0, %1, off"
               :: "v"(lds_addr), "v"((uint64_t)(size_t)gptr)
               : "memory");
}
__device__ __forceinline__ void wait_asynccnt0() {
  asm volatile("s_wait_asynccnt 0x0" ::: "memory");
}

// Load a 16x32 (A) or 32x16 (B) bf16 WMMA fragment from row-major [row][k] storage.
// ISA 7.12.2: lane<16 holds K = {k0..k0+7, k0+16..k0+23}; lane>=16 same +8.
__device__ __forceinline__ v16bf load_frag(const __bf16* base, int stride, int row,
                                           int k0, int lane) {
  const int half = (lane >> 4) & 1;
  const __bf16* p = base + row * stride + k0 + half * 8;
  v8bf lo = *(const v8bf*)(p);
  v8bf hi = *(const v8bf*)(p + 16);
  v16bf f;
#pragma unroll
  for (int i = 0; i < 8; ++i) { f[i] = lo[i]; f[i + 8] = hi[i]; }
  return f;
}

__device__ __forceinline__ v8f wmma_bf16(v16bf a, v16bf b, v8f c) {
  return __builtin_amdgcn_wmma_f32_16x16x32_bf16(false, a, false, b, (short)0, c,
                                                 false, false);
}

// ---------------- Kernel 1: QKV projection (per-head 64x64 GEMMs), f32 -> bf16 ----
// grid = (NL/64, H, 3), block = 128 (4 waves). Wave w computes rows [rb+16w, +16).
// z=0: Q = query @ Wq ; z=1: K = values @ Wk (reference bug: keys ignored) ; z=2: V = values @ Wv.
__global__ __launch_bounds__(128) void qkv_proj_kernel(
    const float* __restrict__ query, const float* __restrict__ values,
    const float* __restrict__ Wq, const float* __restrict__ Wk,
    const float* __restrict__ Wv, __bf16* __restrict__ qws,
    __bf16* __restrict__ kws, __bf16* __restrict__ vws) {
  __shared__ __bf16 Alds[64 * LDP];
  __shared__ __bf16 Wt[64 * LDP];  // W transposed: Wt[n][k] = W[k][n]
  const int tid = threadIdx.x, lane = tid & 31, wave = tid >> 5;
  const int half = (lane >> 4) & 1;
  const int rb = blockIdx.x * 64;
  const int h = blockIdx.y;
  const int which = blockIdx.z;
  const float* in = (which == 0) ? query : values;
  const float* W = (which == 0) ? Wq : (which == 1) ? Wk : Wv;
  __bf16* out = (which == 0) ? qws : (which == 1) ? kws : vws;

#pragma unroll
  for (int i = 0; i < 32; ++i) {  // stage A tile (64 rows x 64 d), coalesced
    int idx = i * 128 + tid;
    int r = idx >> 6, d = idx & 63;
    Alds[r * LDP + d] = (__bf16)in[(size_t)(rb + r) * E_ + h * D_ + d];
  }
#pragma unroll
  for (int i = 0; i < 32; ++i) {  // stage W transposed
    int idx = i * 128 + tid;
    int k = idx >> 6, n = idx & 63;
    Wt[n * LDP + k] = (__bf16)W[k * 64 + n];
  }
  __syncthreads();

  v16bf af[2];
#pragma unroll
  for (int ks = 0; ks < 2; ++ks)
    af[ks] = load_frag(Alds, LDP, wave * 16 + (lane & 15), ks * 32, lane);

  __bf16* Aw = Alds + wave * 16 * LDP;  // wave-private staging (A rows already consumed)
#pragma unroll
  for (int s = 0; s < 4; ++s) {
    v8f c = {};
#pragma unroll
    for (int ks = 0; ks < 2; ++ks) {
      v16bf b = load_frag(Wt, LDP, s * 16 + (lane & 15), ks * 32, lane);
      c = wmma_bf16(af[ks], b, c);
    }
#pragma unroll
    for (int r = 0; r < 8; ++r)  // C layout: M = r + 8*lane[4]
      Aw[(r + 8 * half) * LDP + s * 16 + (lane & 15)] = (__bf16)c[r];
  }
  // coalesced writeback: 2 lanes per row, 4 x b128 per lane
  {
    const int row_l = lane >> 1;
    const int doff = (lane & 1) * 32;
    const int row = rb + wave * 16 + row_l;
    const int n = row / L_, l = row & (L_ - 1);
    __bf16* dst = out + (size_t)((n * H_ + h) * L_ + l) * D_ + doff;
#pragma unroll
    for (int j = 0; j < 4; ++j)
      *(v8bf*)(dst + j * 8) = *(const v8bf*)&Aw[row_l * LDP + doff + j * 8];
  }
}

// ---------------- Kernel 2: flash attention per (n,h) -----------------------------
// grid = (L/64, H, N), block = 128 (4 waves). Each wave owns 16 Q rows; block
// streams K/V in 64-row tiles with double-buffered LDS: K arrives via CDNA5 async
// global->LDS copies (ASYNCcnt), V is register-staged + transposed; both for tile
// j+1 overlap the 16 WMMAs + online softmax of tile j.
__global__ __launch_bounds__(128) void flash_attn_kernel(
    const __bf16* __restrict__ qws, const __bf16* __restrict__ kws,
    const __bf16* __restrict__ vws, __bf16* __restrict__ aws) {
  __shared__ __bf16 Klds[2][64 * LDP];      // [buf][kv][d]
  __shared__ __bf16 Vt[2][64 * LDP];        // [buf][d][kv] (transposed)
  __shared__ __bf16 Plds[4 * 16 * LDP];     // wave-private staging [16 q][64 kv]
  const int tid = threadIdx.x, lane = tid & 31, wave = tid >> 5;
  const int half = (lane >> 4) & 1;
  const int qb = blockIdx.x * 64;
  const int h = blockIdx.y, n = blockIdx.z;
  const float scale = 0.04419417382415922f;  // 1/sqrt(512) per reference

  const size_t head_off = (size_t)(n * H_ + h) * L_ * D_;
  const __bf16* Q = qws + head_off;
  const __bf16* K = kws + head_off;
  const __bf16* V = vws + head_off;
  __bf16* Pw = Plds + wave * 16 * LDP;

  v16bf qf[2];
#pragma unroll
  for (int ks = 0; ks < 2; ++ks)
    qf[ks] = load_frag(Q + (size_t)qb * D_, D_, wave * 16 + (lane & 15), ks * 32, lane);

  v8f Oacc[4] = {};
  float m_i[8], l_i[8];
#pragma unroll
  for (int r = 0; r < 8; ++r) { m_i[r] = -1e30f; l_i[r] = 0.f; }

  // -------- prologue: stage tile 0 into buffer 0 --------
#pragma unroll
  for (int i = 0; i < 4; ++i) {
    int idx = (i * 128 + tid) * 8;
    int r = idx >> 6, d = idx & 63;
    async_load_b128(lds_offset(&Klds[0][r * LDP + d]), &K[idx]);
  }
#pragma unroll
  for (int i = 0; i < 4; ++i) {
    int idx = (i * 128 + tid) * 8;
    int r = idx >> 6, d0 = idx & 63;
    v8bf tmp = *(const v8bf*)&V[idx];
#pragma unroll
    for (int e = 0; e < 8; ++e) Vt[0][(d0 + e) * LDP + r] = tmp[e];
  }
  wait_asynccnt0();
  __syncthreads();

  for (int jt = 0; jt < L_ / 64; ++jt) {
    const int cur = jt & 1, nxt = cur ^ 1;
    const bool more = (jt + 1) < (L_ / 64);
    v8bf vreg[4];
    if (more) {  // kick off tile j+1 before computing on tile j
      const size_t kb1 = (size_t)(jt + 1) * 64 * D_;
#pragma unroll
      for (int i = 0; i < 4; ++i) {
        int idx = (i * 128 + tid) * 8;
        int r = idx >> 6, d = idx & 63;
        async_load_b128(lds_offset(&Klds[nxt][r * LDP + d]), &K[kb1 + idx]);
      }
#pragma unroll
      for (int i = 0; i < 4; ++i) {
        int idx = (i * 128 + tid) * 8;
        vreg[i] = *(const v8bf*)&V[kb1 + idx];
      }
    }

    // S = Q K^T  (16 q x 64 kv per wave, f32 accum)
    v8f sacc[4] = {};
#pragma unroll
    for (int s = 0; s < 4; ++s)
#pragma unroll
      for (int ks = 0; ks < 2; ++ks) {
        v16bf b = load_frag(&Klds[cur][0], LDP, s * 16 + (lane & 15), ks * 32, lane);
        sacc[s] = wmma_bf16(qf[ks], b, sacc[s]);
      }

    // online softmax; row r lives on 16 lanes of one half-wave -> xor-shuffle reduce
#pragma unroll
    for (int r = 0; r < 8; ++r) {
      float vmax = fmaxf(fmaxf(sacc[0][r], sacc[1][r]), fmaxf(sacc[2][r], sacc[3][r]));
#pragma unroll
      for (int off = 1; off < 16; off <<= 1)
        vmax = fmaxf(vmax, __shfl_xor(vmax, off, 32));
      float mnew = fmaxf(m_i[r], vmax);
      float alpha = __expf(scale * (m_i[r] - mnew));
      m_i[r] = mnew;
      float psum = 0.f;
#pragma unroll
      for (int s = 0; s < 4; ++s) {
        float p = __expf(scale * (sacc[s][r] - mnew));
        sacc[s][r] = p;
        psum += p;
      }
#pragma unroll
      for (int off = 1; off < 16; off <<= 1) psum += __shfl_xor(psum, off, 32);
      l_i[r] = l_i[r] * alpha + psum;
#pragma unroll
      for (int s = 0; s < 4; ++s) Oacc[s][r] *= alpha;
    }

    // P: C-layout -> wave-private LDS -> A-layout fragments
#pragma unroll
    for (int s = 0; s < 4; ++s)
#pragma unroll
      for (int r = 0; r < 8; ++r)
        Pw[(r + 8 * half) * LDP + s * 16 + (lane & 15)] = (__bf16)sacc[s][r];

    v16bf pf0 = load_frag(Pw, LDP, lane & 15, 0, lane);
    v16bf pf1 = load_frag(Pw, LDP, lane & 15, 32, lane);

    // O += P V   (B fragments from transposed V: contiguous kv runs)
#pragma unroll
    for (int s = 0; s < 4; ++s) {
      v16bf b0 = load_frag(&Vt[cur][0], LDP, s * 16 + (lane & 15), 0, lane);
      Oacc[s] = wmma_bf16(pf0, b0, Oacc[s]);
      v16bf b1 = load_frag(&Vt[cur][0], LDP, s * 16 + (lane & 15), 32, lane);
      Oacc[s] = wmma_bf16(pf1, b1, Oacc[s]);
    }

    if (more) {  // drain tile j+1 staging into the spare buffer
#pragma unroll
      for (int i = 0; i < 4; ++i) {
        int idx = (i * 128 + tid) * 8;
        int r = idx >> 6, d0 = idx & 63;
#pragma unroll
        for (int e = 0; e < 8; ++e) Vt[nxt][(d0 + e) * LDP + r] = vreg[i][e];
      }
      wait_asynccnt0();
    }
    __syncthreads();
  }

  // epilogue: normalize, stage in wave-private LDS, coalesced b128 stores
#pragma unroll
  for (int s = 0; s < 4; ++s)
#pragma unroll
    for (int r = 0; r < 8; ++r)
      Pw[(r + 8 * half) * LDP + s * 16 + (lane & 15)] =
          (__bf16)(Oacc[s][r] / l_i[r]);
  {
    const int row_l = lane >> 1;
    const int doff = (lane & 1) * 32;
    const int row = qb + wave * 16 + row_l;
    __bf16* dst = aws + (size_t)(n * L_ + row) * E_ + h * D_ + doff;
#pragma unroll
    for (int j = 0; j < 4; ++j)
      *(v8bf*)(dst + j * 8) = *(const v8bf*)&Pw[row_l * LDP + doff + j * 8];
  }
}

// ---------------- Kernel 3: output projection [NL,512] x [512,512] + bias ---------
// grid = (NL/64, E/64), block = 128 (4 waves). K streamed in 64-chunks through a
// double-buffered LDS tile of Wo^T with register staging (one barrier per chunk).
__global__ __launch_bounds__(128) void out_proj_kernel(
    const __bf16* __restrict__ X, const float* __restrict__ Wo,
    const float* __restrict__ bo, float* __restrict__ out) {
  __shared__ __bf16 Wt[2][64 * LDP];  // Wo chunk transposed: Wt[buf][n][k]
  const int tid = threadIdx.x, lane = tid & 31, wave = tid >> 5;
  const int half = (lane >> 4) & 1;
  const int rb = blockIdx.x * 64;
  const int cb = blockIdx.y * 64;
  const int NCH = E_ / 64;

  // prologue: stage chunk 0 into buffer 0
#pragma unroll
  for (int i = 0; i < 8; ++i) {
    int idx = (i * 128 + tid) * 4;
    int k = idx >> 6, nn = idx & 63;
    float4 w = *(const float4*)&Wo[(size_t)k * E_ + cb + nn];
    Wt[0][(nn + 0) * LDP + k] = (__bf16)w.x;
    Wt[0][(nn + 1) * LDP + k] = (__bf16)w.y;
    Wt[0][(nn + 2) * LDP + k] = (__bf16)w.z;
    Wt[0][(nn + 3) * LDP + k] = (__bf16)w.w;
  }
  __syncthreads();

  v8f acc[4] = {};
  for (int c = 0; c < NCH; ++c) {
    const int cur = c & 1, nxt = cur ^ 1;
    const bool more = (c + 1) < NCH;
    const int k0 = c * 64;
    float4 wreg[8];
    if (more) {  // fetch next Wo chunk into registers, overlapping the WMMAs
#pragma unroll
      for (int i = 0; i < 8; ++i) {
        int idx = (i * 128 + tid) * 4;
        int k = idx >> 6, nn = idx & 63;
        wreg[i] = *(const float4*)&Wo[(size_t)(k0 + 64 + k) * E_ + cb + nn];
      }
    }
    v16bf a0 = load_frag(X, E_, rb + wave * 16 + (lane & 15), k0, lane);
    v16bf a1 = load_frag(X, E_, rb + wave * 16 + (lane & 15), k0 + 32, lane);
#pragma unroll
    for (int s = 0; s < 4; ++s) {
      v16bf b0 = load_frag(&Wt[cur][0], LDP, s * 16 + (lane & 15), 0, lane);
      acc[s] = wmma_bf16(a0, b0, acc[s]);
      v16bf b1 = load_frag(&Wt[cur][0], LDP, s * 16 + (lane & 15), 32, lane);
      acc[s] = wmma_bf16(a1, b1, acc[s]);
    }
    if (more) {  // drain next chunk into the spare buffer
#pragma unroll
      for (int i = 0; i < 8; ++i) {
        int idx = (i * 128 + tid) * 4;
        int k = idx >> 6, nn = idx & 63;
        Wt[nxt][(nn + 0) * LDP + k] = (__bf16)wreg[i].x;
        Wt[nxt][(nn + 1) * LDP + k] = (__bf16)wreg[i].y;
        Wt[nxt][(nn + 2) * LDP + k] = (__bf16)wreg[i].z;
        Wt[nxt][(nn + 3) * LDP + k] = (__bf16)wreg[i].w;
      }
    }
    __syncthreads();
  }
#pragma unroll
  for (int s = 0; s < 4; ++s)
#pragma unroll
    for (int r = 0; r < 8; ++r) {
      int row = rb + wave * 16 + r + 8 * half;
      int col = cb + s * 16 + (lane & 15);
      out[(size_t)row * E_ + col] = acc[s][r] + bo[col];
    }
}

// ---------------- Launch -----------------------------------------------------------
extern "C" void kernel_launch(void* const* d_in, const int* in_sizes, int n_in,
                              void* d_out, int out_size, void* d_ws, size_t ws_size,
                              hipStream_t stream) {
  (void)in_sizes; (void)n_in; (void)out_size; (void)ws_size;
  const float* values = (const float*)d_in[0];
  const float* keys   = (const float*)d_in[1]; (void)keys;  // reference ignores keys
  const float* query  = (const float*)d_in[2];
  // d_in[3] = masks: all-ones and unused by the reference math
  const float* Wv = (const float*)d_in[4];
  const float* Wk = (const float*)d_in[5];
  const float* Wq = (const float*)d_in[6];
  const float* Wo = (const float*)d_in[7];
  const float* bo = (const float*)d_in[8];

  // workspace: Q | K | V | attn-out, each NL*E bf16  (total 32 MiB)
  __bf16* qws = (__bf16*)d_ws;
  __bf16* kws = qws + NLE_;
  __bf16* vws = kws + NLE_;
  __bf16* aws = vws + NLE_;

  qkv_proj_kernel<<<dim3(NL_ / 64, H_, 3), 128, 0, stream>>>(query, values, Wq, Wk,
                                                             Wv, qws, kws, vws);
  flash_attn_kernel<<<dim3(L_ / 64, H_, N_), 128, 0, stream>>>(qws, kws, vws, aws);
  out_proj_kernel<<<dim3(NL_ / 64, E_ / 64), 128, 0, stream>>>(aws, Wo, bo,
                                                               (float*)d_out);
}
